// EdgeGNN_17162689315092
// MI455X (gfx1250) — compile-verified, matrix operand order checked by reading, not run on hardware
//
#include <hip/hip_runtime.h>
#include <hip/hip_bf16.h>
#include <cstddef>

// ---------------------------------------------------------------------------
// Problem constants (from reference)
// ---------------------------------------------------------------------------
namespace {
constexpr int Bn  = 64;          // batches
constexpr int NPn = 512;         // positions per batch (L)
constexpr int Hh  = 128;         // hidden
constexpr int DSn = 16;          // state size
constexpr int DIn = 256;         // inner dim
constexpr int DRn = 8;           // dt rank
constexpr int Nn  = Bn * NPn;    // 32768 nodes
constexpr int En  = 262144;      // edges
constexpr int CE  = 65536;       // edge chunk for classifier
constexpr int DBLP = 48;         // padded dbl row stride (was 40)
constexpr float EPSc   = 1e-5f;
constexpr float SLOPEc = 0.01f;
}

typedef __bf16 bf16_t;
typedef bf16_t v16bf __attribute__((ext_vector_type(16)));
typedef bf16_t v8bf  __attribute__((ext_vector_type(8)));
typedef float  v8f   __attribute__((ext_vector_type(8)));
typedef float  f32x4 __attribute__((ext_vector_type(4)));

__device__ __forceinline__ bf16_t cvt_bf16(float f) { return (bf16_t)f; }

__device__ __forceinline__ float act_apply(float v, int act) {
    if (act == 1) return v > 0.f ? v : 0.f;                        // relu
    if (act == 2) return v > 0.f ? v : SLOPEc * v;                 // leaky
    if (act == 3) return (v > 20.f) ? v : __logf(1.f + __expf(v)); // softplus
    return v;
}

// ---------------------------------------------------------------------------
// Weight pre-pack: W[Kd x Nc] (row stride ldw) -> Wt[Np][Kp] bf16,
// transposed + zero padded so the GEMM hot loop needs no guards.
// ---------------------------------------------------------------------------
__global__ void __launch_bounds__(256)
pack_wt_kernel(const float* __restrict__ W, int ldw, int Kd, int Nc,
               bf16_t* __restrict__ out, int Kp, int Np)
{
    int i = blockIdx.x * blockDim.x + threadIdx.x;
    if (i >= Np * Kp) return;
    int c = i / Kp;
    int k = i % Kp;
    float v = (c < Nc && k < Kd) ? W[(size_t)k * ldw + c] : 0.f;
    out[i] = cvt_bf16(v);
}

__global__ void pad_bias_kernel(const float* __restrict__ b,
                                float* __restrict__ out, int n, int np)
{
    int i = blockIdx.x * blockDim.x + threadIdx.x;
    if (i < np) out[i] = (i < n) ? b[i] : 0.f;
}

// ---------------------------------------------------------------------------
// bf16-WMMA GEMM: C[M x Np] = act(A[M x Kp] * Wt^T + bias) (+C if acc)
//   A: f32, row stride lda (>= Kp, 32B-aligned rows), converted in-register.
//   Wt: bf16 packed [Np][Kp] (zero padded) -> contiguous 16B fragment loads.
//   One wave computes a 32x16 tile (2 WMMA / k-step, B fragment reused).
//   No guards anywhere in the hot loop; EXEC stays all-ones for WMMA.
// ---------------------------------------------------------------------------
__global__ void __launch_bounds__(256)
gemm_wmma_kernel(const float* __restrict__ A, int lda,
                 const bf16_t* __restrict__ Wt,
                 float* __restrict__ C, int ldc,
                 int M, int Kp, int Np,
                 const float* __restrict__ bias, int act, int acc)
{
    int wave = (int)((blockIdx.x * (unsigned)blockDim.x + threadIdx.x) >> 5);
    int tnc  = Np >> 4;
    int tmc  = M >> 5;                       // 32 rows per wave
    if (wave >= tmc * tnc) return;           // wave-uniform exit
    int tm = wave / tnc;
    int tn = wave % tnc;

    int lane = threadIdx.x & 31;
    int hi   = lane >> 4;                    // lane half
    int rc   = lane & 15;
    int kofs = hi ? 8 : 0;

    const float*  arow0 = A  + (size_t)(tm * 32 + rc) * lda;
    const float*  arow1 = arow0 + (size_t)16 * lda;
    const bf16_t* wcol  = Wt + (size_t)(tn * 16 + rc) * Kp;

    v8f c0 = {};
    v8f c1 = {};
    for (int k0 = 0; k0 < Kp; k0 += 32) {
        int kb = k0 + kofs;
        __builtin_prefetch(arow0 + k0 + 64, 0, 0);
        // A fragments: two contiguous 8-float runs per lane -> 4x b128 each
        f32x4 a00 = *(const f32x4*)(arow0 + kb);
        f32x4 a01 = *(const f32x4*)(arow0 + kb + 4);
        f32x4 a02 = *(const f32x4*)(arow0 + kb + 16);
        f32x4 a03 = *(const f32x4*)(arow0 + kb + 20);
        f32x4 a10 = *(const f32x4*)(arow1 + kb);
        f32x4 a11 = *(const f32x4*)(arow1 + kb + 4);
        f32x4 a12 = *(const f32x4*)(arow1 + kb + 16);
        f32x4 a13 = *(const f32x4*)(arow1 + kb + 20);
        // B fragment: two 16B bf16 runs from the packed weight
        v8bf blo = *(const v8bf*)(wcol + kb);
        v8bf bhi = *(const v8bf*)(wcol + kb + 16);
        v16bf b  = __builtin_shufflevector(blo, bhi,
                       0, 1, 2, 3, 4, 5, 6, 7, 8, 9, 10, 11, 12, 13, 14, 15);
        v16bf a0, a1;
#pragma unroll
        for (int i = 0; i < 4; ++i) {
            a0[i]      = cvt_bf16(a00[i]);
            a0[4 + i]  = cvt_bf16(a01[i]);
            a0[8 + i]  = cvt_bf16(a02[i]);
            a0[12 + i] = cvt_bf16(a03[i]);
            a1[i]      = cvt_bf16(a10[i]);
            a1[4 + i]  = cvt_bf16(a11[i]);
            a1[8 + i]  = cvt_bf16(a12[i]);
            a1[12 + i] = cvt_bf16(a13[i]);
        }
        c0 = __builtin_amdgcn_wmma_f32_16x16x32_bf16(false, a0, false, b,
                                                     (short)0, c0, false, false);
        c1 = __builtin_amdgcn_wmma_f32_16x16x32_bf16(false, a1, false, b,
                                                     (short)0, c1, false, false);
    }

    int bcol = tn * 16 + rc;
    float bv = (bias != nullptr) ? bias[bcol] : 0.f;
    int rbase = tm * 32 + (hi ? 8 : 0);
#pragma unroll
    for (int r = 0; r < 8; ++r) {
        size_t off0 = (size_t)(rbase + r) * ldc + bcol;
        size_t off1 = (size_t)(rbase + 16 + r) * ldc + bcol;
        float v0 = act_apply(c0[r] + bv, act);
        float v1 = act_apply(c1[r] + bv, act);
        if (acc) { v0 += C[off0]; v1 += C[off1]; }
        C[off0] = v0;
        C[off1] = v1;
    }
}

// ---------------------------------------------------------------------------
// Batchnorm stats (per-column mean/var over rows)
// ---------------------------------------------------------------------------
__global__ void __launch_bounds__(256)
colstats_kernel(const float* __restrict__ x, int nrows, int width,
                float* __restrict__ mean, float* __restrict__ var)
{
    __shared__ float ss[256];
    __shared__ float sq[256];
    int col = blockIdx.x;
    float s = 0.f, q = 0.f;
    for (int r = threadIdx.x; r < nrows; r += blockDim.x) {
        float v = x[(size_t)r * width + col];
        s += v; q += v * v;
    }
    ss[threadIdx.x] = s; sq[threadIdx.x] = q;
    __syncthreads();
    for (int ofs = 128; ofs > 0; ofs >>= 1) {
        if (threadIdx.x < ofs) {
            ss[threadIdx.x] += ss[threadIdx.x + ofs];
            sq[threadIdx.x] += sq[threadIdx.x + ofs];
        }
        __syncthreads();
    }
    if (threadIdx.x == 0) {
        float m = ss[0] / (float)nrows;
        mean[col] = m;
        var[col]  = sq[0] / (float)nrows - m * m;
    }
}

__global__ void __launch_bounds__(256)
bn_apply_kernel(const float* __restrict__ x, const float* __restrict__ mean,
                const float* __restrict__ var, const float* __restrict__ g,
                const float* __restrict__ b, float* __restrict__ out,
                size_t n, int width)
{
    size_t i = (size_t)blockIdx.x * blockDim.x + threadIdx.x;
    if (i >= n) return;
    int c = (int)(i % (size_t)width);
    out[i] = (x[i] - mean[c]) * rsqrtf(var[c] + EPSc) * g[c] + b[c];
}

// ---------------------------------------------------------------------------
// Elementwise helpers
// ---------------------------------------------------------------------------
__global__ void __launch_bounds__(256)
zero_kernel(float* __restrict__ p, size_t n)
{
    size_t i = (size_t)blockIdx.x * blockDim.x + threadIdx.x;
    if (i < n) p[i] = 0.f;
}

__global__ void __launch_bounds__(256)
add_kernel(const float* __restrict__ a, const float* __restrict__ b,
           float* __restrict__ o, size_t n)
{
    size_t i = (size_t)blockIdx.x * blockDim.x + threadIdx.x;
    if (i < n) o[i] = a[i] + b[i];
}

// msg = relu(xn[src] + (ea@lin_w + lin_b)); scatter-add into agg[dst]
__global__ void __launch_bounds__(256)
edge_msg_kernel(const float* __restrict__ t, const float* __restrict__ xn,
                const int* __restrict__ src, const int* __restrict__ dst,
                float* __restrict__ agg)
{
    size_t i = (size_t)blockIdx.x * blockDim.x + threadIdx.x;
    if (i >= (size_t)En * Hh) return;
    int e = (int)(i >> 7);
    int c = (int)(i & 127);
    float v = xn[(size_t)src[e] * Hh + c] + t[i];
    v = v > 0.f ? v : 0.f;
    atomicAdd(&agg[(size_t)dst[e] * Hh + c], v);
}

// Extract the 2 real output columns from the padded classifier result
__global__ void __launch_bounds__(256)
copy_out2_kernel(const float* __restrict__ Cpad, float* __restrict__ out, int e0)
{
    int i = blockIdx.x * blockDim.x + threadIdx.x;
    if (i >= CE * 2) return;
    int e = i >> 1;
    int c = i & 1;
    out[(size_t)(e0 + e) * 2 + c] = Cpad[(size_t)e * 16 + c];
}

// ---------------------------------------------------------------------------
// Row layernorm: one block per row, blockDim == width (128 or 256, pow2)
// ---------------------------------------------------------------------------
__global__ void rownorm_kernel(const float* __restrict__ in,
                               const float* __restrict__ g,
                               const float* __restrict__ b,
                               float* __restrict__ out, int width)
{
    __shared__ float red[256];
    __shared__ float mv[2];
    int row = blockIdx.x;
    int c   = threadIdx.x;
    float v = in[(size_t)row * width + c];
    red[c] = v;
    __syncthreads();
    for (int ofs = width >> 1; ofs > 0; ofs >>= 1) {
        if (c < ofs) red[c] += red[c + ofs];
        __syncthreads();
    }
    if (c == 0) mv[0] = red[0] / (float)width;
    __syncthreads();
    float d = v - mv[0];
    red[c] = d * d;
    __syncthreads();
    for (int ofs = width >> 1; ofs > 0; ofs >>= 1) {
        if (c < ofs) red[c] += red[c + ofs];
        __syncthreads();
    }
    if (c == 0) mv[1] = rsqrtf(red[0] / (float)width + EPSc);
    __syncthreads();
    out[(size_t)row * width + c] = d * mv[1] * g[c] + b[c];
}

// er = concat(h[src], h[dst]) row, then layernorm(c_ln1) -> out (chunk-local)
__global__ void __launch_bounds__(256)
edge_concat_ln_kernel(const float* __restrict__ h, const int* __restrict__ src,
                      const int* __restrict__ dst, const float* __restrict__ g,
                      const float* __restrict__ b, float* __restrict__ out, int e0)
{
    __shared__ float red[256];
    __shared__ float mv[2];
    int e = e0 + blockIdx.x;
    int c = threadIdx.x;
    float v = (c < Hh) ? h[(size_t)src[e] * Hh + c]
                       : h[(size_t)dst[e] * Hh + (c - Hh)];
    red[c] = v;
    __syncthreads();
    for (int ofs = 128; ofs > 0; ofs >>= 1) {
        if (c < ofs) red[c] += red[c + ofs];
        __syncthreads();
    }
    if (c == 0) mv[0] = red[0] / 256.f;
    __syncthreads();
    float d = v - mv[0];
    red[c] = d * d;
    __syncthreads();
    for (int ofs = 128; ofs > 0; ofs >>= 1) {
        if (c < ofs) red[c] += red[c + ofs];
        __syncthreads();
    }
    if (c == 0) mv[1] = rsqrtf(red[0] / 256.f + EPSc);
    __syncthreads();
    out[(size_t)blockIdx.x * 256 + c] = d * mv[1] * g[c] + b[c];
}

// ---------------------------------------------------------------------------
// Causal depthwise conv (K=4) over xi = xz[:, :256], then SiLU -> xc
// ---------------------------------------------------------------------------
__global__ void __launch_bounds__(256)
conv_silu_kernel(const float* __restrict__ xz, const float* __restrict__ cw,
                 const float* __restrict__ cb, float* __restrict__ xc)
{
    size_t i = (size_t)blockIdx.x * blockDim.x + threadIdx.x;
    if (i >= (size_t)Nn * DIn) return;
    int d = (int)(i & 255);
    size_t n = i >> 8;
    int p = (int)(n & 511);
    size_t bb = n >> 9;
    float acc = cb[d];
#pragma unroll
    for (int k = 0; k < 4; ++k) {
        int pp = p + k - 3;
        if (pp >= 0)
            acc += xz[((bb << 9) + (size_t)pp) * (2 * DIn) + d] * cw[d * 4 + k];
    }
    float sg = 1.f / (1.f + __expf(-acc));
    xc[i] = acc * sg;
}

// ---------------------------------------------------------------------------
// Selective scan: one block per batch, one thread per channel d.
// 16-wide state in registers; B/C (shared across channels) staged in LDS.
// Fused: y = (scan + Dm*xc) * silu(z), written to ybuf.
// ---------------------------------------------------------------------------
__global__ void __launch_bounds__(256)
scan_kernel(const float* __restrict__ dt, const float* __restrict__ xc,
            const float* __restrict__ dbl, const float* __restrict__ A_log,
            const float* __restrict__ Dm, const float* __restrict__ xz,
            float* __restrict__ y)
{
    __shared__ float sB[DSn];
    __shared__ float sC[DSn];
    int d  = threadIdx.x;
    int bb = blockIdx.x;

    float a[DSn];
    float hs[DSn];
#pragma unroll
    for (int s = 0; s < DSn; ++s) {
        a[s]  = -__expf(A_log[(size_t)d * DSn + s]);
        hs[s] = 0.f;
    }
    float Dd = Dm[d];

    for (int p = 0; p < NPn; ++p) {
        size_t n = (size_t)bb * NPn + p;
        if (threadIdx.x < 32) {
            int s = threadIdx.x & 15;
            float v = dbl[n * DBLP + DRn + ((threadIdx.x < 16) ? s : DSn + s)];
            if (threadIdx.x < 16) sB[s] = v; else sC[s] = v;
        }
        __syncthreads();
        float dtv = dt[n * DIn + d];
        float xcv = xc[n * DIn + d];
        float yac = 0.f;
#pragma unroll
        for (int s = 0; s < DSn; ++s) {
            float dA = __expf(dtv * a[s]);
            hs[s] = dA * hs[s] + dtv * sB[s] * xcv;
            yac  += hs[s] * sC[s];
        }
        float zv  = xz[n * (2 * DIn) + DIn + d];
        float sg  = 1.f / (1.f + __expf(-zv));
        y[n * DIn + d] = (yac + Dd * xcv) * (zv * sg);
        __syncthreads();
    }
}

// ---------------------------------------------------------------------------
// Host-side helpers
// ---------------------------------------------------------------------------
static inline void gemm(const float* A, int lda, const bf16_t* Wt,
                        float* C, int ldc, int M, int Kp, int Np,
                        const float* bias, int act, int acc, hipStream_t s)
{
    long tiles = (long)(M >> 5) * (Np >> 4);
    unsigned blocks = (unsigned)((tiles * 32 + 255) / 256);
    gemm_wmma_kernel<<<blocks, 256, 0, s>>>(A, lda, Wt, C, ldc,
                                            M, Kp, Np, bias, act, acc);
}

static inline void pack_wt(const float* W, int ldw, int Kd, int Nc,
                           bf16_t* out, int Kp, int Np, hipStream_t s)
{
    int n = Np * Kp;
    pack_wt_kernel<<<(n + 255) / 256, 256, 0, s>>>(W, ldw, Kd, Nc, out, Kp, Np);
}

static inline unsigned nblk(size_t n) { return (unsigned)((n + 255) / 256); }

extern "C" void kernel_launch(void* const* d_in, const int* in_sizes, int n_in,
                              void* d_out, int out_size, void* d_ws, size_t ws_size,
                              hipStream_t stream)
{
    (void)in_sizes; (void)n_in; (void)out_size; (void)ws_size;

    const float* x      = (const float*)d_in[0];
    const int*   ei     = (const int*)  d_in[1];
    const float* eattr  = (const float*)d_in[2];
    const float* bng    = (const float*)d_in[3];
    const float* bnb    = (const float*)d_in[4];
    const float* beg    = (const float*)d_in[5];
    const float* beb    = (const float*)d_in[6];
    const float* lin_w  = (const float*)d_in[7];
    const float* lin_b  = (const float*)d_in[8];
    const float* w1     = (const float*)d_in[9];
    const float* b1     = (const float*)d_in[10];
    const float* w2     = (const float*)d_in[11];
    const float* b2     = (const float*)d_in[12];
    const float* mlng   = (const float*)d_in[13];
    const float* mlnb   = (const float*)d_in[14];
    const float* w_in   = (const float*)d_in[15];
    const float* conv_w = (const float*)d_in[16];
    const float* conv_b = (const float*)d_in[17];
    const float* w_x    = (const float*)d_in[18];
    const float* w_dt   = (const float*)d_in[19];
    const float* b_dt   = (const float*)d_in[20];
    const float* A_log  = (const float*)d_in[21];
    const float* Dm     = (const float*)d_in[22];
    const float* w_out  = (const float*)d_in[23];
    const float* c1g    = (const float*)d_in[24];
    const float* c1b    = (const float*)d_in[25];
    const float* cw1    = (const float*)d_in[26];
    const float* cb1    = (const float*)d_in[27];
    const float* c2g    = (const float*)d_in[28];
    const float* c2b    = (const float*)d_in[29];
    const float* cw2    = (const float*)d_in[30];
    const float* cb2    = (const float*)d_in[31];
    const float* c3g    = (const float*)d_in[32];
    const float* c3b    = (const float*)d_in[33];
    const float* cw3    = (const float*)d_in[34];
    const float* cb3    = (const float*)d_in[35];

    float* ws = (float*)d_ws;
    size_t o = 0;
    auto take = [&](size_t nfl) { float* p = ws + o; o += nfl; return p; };
    float* meanx = take(Hh);  float* varx = take(Hh);
    float* meane = take(Hh);  float* vare = take(Hh);
    float* cb3p  = take(16);
    float* xn    = take((size_t)Nn * Hh);
    float* agg   = take((size_t)Nn * Hh);
    float* hbuf  = take((size_t)Nn * Hh);
    float* hn    = take((size_t)Nn * Hh);
    float* ean   = take((size_t)En * Hh);          // dead after msg phase
    float* tedg  = take((size_t)En * Hh);          // dead after msg phase
    float* xz    = take((size_t)Nn * (2 * DIn));   // dead after mamba
    float* xcb   = take((size_t)Nn * DIn);
    float* dblb  = take((size_t)Nn * DBLP);        // padded stride 48
    float* dtb   = take((size_t)Nn * DIn);
    float* ybuf  = take((size_t)Nn * DIn);

    // packed bf16 weights (carved from the float pool; counts are even)
    auto takeb = [&](size_t nbf) { return (bf16_t*)take(nbf / 2); };
    bf16_t* p_lin  = takeb(128 * 128);
    bf16_t* p_w1   = takeb(128 * 128);
    bf16_t* p_w2   = takeb(128 * 128);
    bf16_t* p_win  = takeb(128 * 512);
    bf16_t* p_wx   = takeb(256 * 48);
    bf16_t* p_wdt  = takeb(32 * 256);
    bf16_t* p_wout = takeb(256 * 128);
    bf16_t* p_cw1  = takeb(256 * 256);
    bf16_t* p_cw2  = takeb(256 * 128);
    bf16_t* p_cw3  = takeb(128 * 16);

    // classifier chunk buffers alias dead regions:
    float* Z1   = xz;                        // CE*256 == Nn*512 floats
    float* Z2   = ean;                       // CE*256 <= En*128 floats
    float* Z3   = tedg;                      // CE*128
    float* Z3b  = tedg + (size_t)CE * Hh;    // CE*128
    float* Cpad = tedg + (size_t)2 * CE * Hh;// CE*16

    const int* src = ei;
    const int* dst = ei + En;
    float* out = (float*)d_out;

    // ---- 0. pack all GEMM weights to padded transposed bf16 -------------
    pack_wt(lin_w, 128, 128, 128, p_lin,  128, 128, stream);
    pack_wt(w1,    128, 128, 128, p_w1,   128, 128, stream);
    pack_wt(w2,    128, 128, 128, p_w2,   128, 128, stream);
    pack_wt(w_in,  512, 128, 512, p_win,  128, 512, stream);
    pack_wt(w_x,    40, 256,  40, p_wx,   256,  48, stream);
    pack_wt(w_dt,  256,   8, 256, p_wdt,   32, 256, stream);  // zero K-pad 8->32
    pack_wt(w_out, 128, 256, 128, p_wout, 256, 128, stream);
    pack_wt(cw1,   256, 256, 256, p_cw1,  256, 256, stream);
    pack_wt(cw2,   128, 256, 128, p_cw2,  256, 128, stream);
    pack_wt(cw3,     2, 128,   2, p_cw3,  128,  16, stream);
    pad_bias_kernel<<<1, 16, 0, stream>>>(cb3, cb3p, 2, 16);

    // ---- 1. batchnorm nodes + edges ------------------------------------
    colstats_kernel<<<Hh, 256, 0, stream>>>(x, Nn, Hh, meanx, varx);
    colstats_kernel<<<Hh, 256, 0, stream>>>(eattr, En, Hh, meane, vare);
    bn_apply_kernel<<<nblk((size_t)Nn * Hh), 256, 0, stream>>>(
        x, meanx, varx, bng, bnb, xn, (size_t)Nn * Hh, Hh);
    bn_apply_kernel<<<nblk((size_t)En * Hh), 256, 0, stream>>>(
        eattr, meane, vare, beg, beb, ean, (size_t)En * Hh, Hh);

    // ---- 2. edge message: t = ea@lin_w + lin_b; agg += relu(xn[src]+t) --
    gemm(ean, Hh, p_lin, tedg, Hh, En, 128, 128, lin_b, 0, 0, stream);
    zero_kernel<<<nblk((size_t)Nn * Hh), 256, 0, stream>>>(agg, (size_t)Nn * Hh);
    edge_msg_kernel<<<nblk((size_t)En * Hh), 256, 0, stream>>>(tedg, xn, src, dst, agg);

    // ---- 3. node MLP ----------------------------------------------------
    add_kernel<<<nblk((size_t)Nn * Hh), 256, 0, stream>>>(xn, agg, hbuf, (size_t)Nn * Hh);
    gemm(hbuf, Hh, p_w1, hn, Hh, Nn, 128, 128, b1, 2, 0, stream);     // leaky
    gemm(hn, Hh, p_w2, hbuf, Hh, Nn, 128, 128, b2, 1, 0, stream);     // relu

    // ---- 4. two mamba blocks (residual accumulated in hbuf) ------------
    for (int rep = 0; rep < 2; ++rep) {
        rownorm_kernel<<<Nn, Hh, 0, stream>>>(hbuf, mlng, mlnb, hn, Hh);
        gemm(hn, Hh, p_win, xz, 2 * DIn, Nn, 128, 512, nullptr, 0, 0, stream);
        conv_silu_kernel<<<nblk((size_t)Nn * DIn), 256, 0, stream>>>(
            xz, conv_w, conv_b, xcb);
        gemm(xcb, DIn, p_wx, dblb, DBLP, Nn, 256, 48, nullptr, 0, 0, stream);
        gemm(dblb, DBLP, p_wdt, dtb, DIn, Nn, 32, 256, b_dt, 3, 0, stream); // softplus
        scan_kernel<<<Bn, DIn, 0, stream>>>(dtb, xcb, dblb, A_log, Dm, xz, ybuf);
        gemm(ybuf, DIn, p_wout, hbuf, Hh, Nn, 256, 128, nullptr, 0, 1, stream); // +res
    }

    // ---- 5. edge classifier, chunked over E ----------------------------
    for (int c = 0; c < En / CE; ++c) {
        int e0 = c * CE;
        edge_concat_ln_kernel<<<CE, 256, 0, stream>>>(hbuf, src, dst, c1g, c1b, Z1, e0);
        gemm(Z1, 2 * Hh, p_cw1, Z2, 2 * Hh, CE, 256, 256, cb1, 2, 0, stream); // leaky
        rownorm_kernel<<<CE, 256, 0, stream>>>(Z2, c2g, c2b, Z1, 2 * Hh);
        gemm(Z1, 2 * Hh, p_cw2, Z3, Hh, CE, 256, 128, cb2, 2, 0, stream);     // leaky
        rownorm_kernel<<<CE, Hh, 0, stream>>>(Z3, c3g, c3b, Z3b, Hh);
        gemm(Z3b, Hh, p_cw3, Cpad, 16, CE, 128, 16, cb3p, 0, 0, stream);
        copy_out2_kernel<<<nblk((size_t)CE * 2), 256, 0, stream>>>(Cpad, out, e0);
    }
}
